// NVFP4FakeQuantizedGptOssExperts_16535624090417
// MI455X (gfx1250) — compile-verified, matrix operand order checked by reading, not run on hardware
//
#include <hip/hip_runtime.h>
#include <hip/hip_bf16.h>

#define Tt 512
#define Hh 2880
#define Ii 2880
#define Ee 8

typedef __attribute__((ext_vector_type(16))) _Float16 v16h;
typedef __attribute__((ext_vector_type(8)))  _Float16 v8h;
typedef __attribute__((ext_vector_type(8)))  float    v8f;

static __device__ inline v8f vzero8f() {
  v8f z = {0.f,0.f,0.f,0.f,0.f,0.f,0.f,0.f};
  return z;
}

static __device__ inline void sched_fence() {
#if __has_builtin(__builtin_amdgcn_sched_barrier)
  __builtin_amdgcn_sched_barrier(0);   // keep ds_load burst before the WMMA burst
#endif
}

// ---------- CDNA5 async global->LDS DMA (ASYNCcnt-tracked) ----------
// LDS byte address = low 32 bits of the generic shared pointer (flat LDS aperture
// keeps the offset in addr[31:0]).  Inline asm per CDNA5 ISA.

static __device__ inline void async_ld_b128(unsigned lds_addr, const void* g) {
  unsigned long long ga = (unsigned long long)g;
  asm volatile("global_load_async_to_lds_b128 %0, %1, off"
               :: "v"(lds_addr), "v"(ga) : "memory");
}
static __device__ inline void wait_async0() {
  asm volatile("s_wait_asynccnt 0x0" ::: "memory");
}
static __device__ inline unsigned lds_lo32(const void* p) {
  return (unsigned)(unsigned long long)p;
}

// ---------- NVFP4 fake-quant helpers ----------

// round-to-nearest-even quantization of x>=0 to fp8 e4m3fn values, returned as float
static __device__ inline float quant_e4m3(float x) {
  if (!(x > 0.f)) return 0.f;
  if (x >= 448.f) return 448.f;               // e4m3fn saturates
  unsigned u = __float_as_uint(x);
  int e = (int)((u >> 23) & 0xFF) - 127;
  if (e < -10) return 0.f;                    // below half of min denorm 2^-9
  int qe = (e > -6) ? (e - 3) : -9;           // quantum exponent (3 mantissa bits, denorm floor 2^-9)
  float q = exp2f((float)qe);
  return rintf(x / q) * q;                    // rintf == round-to-nearest-even
}

// e2m1 grid {0,.5,1,1.5,2,3,4,6}, ties to even mantissa index (matches reference searchsorted trick)
static __device__ inline float round_fp4(float x) {
  float a = fminf(fabsf(x), 6.f);
  float q;
  if      (a <= 0.25f) q = 0.0f;
  else if (a <  0.75f) q = 0.5f;
  else if (a <= 1.25f) q = 1.0f;
  else if (a <  1.75f) q = 1.5f;
  else if (a <= 2.5f ) q = 2.0f;
  else if (a <  3.5f ) q = 3.0f;
  else if (a <= 5.0f ) q = 4.0f;
  else                 q = 6.0f;
  return copysignf(q, x);
}

// fake-quant one 16-element block (strided input), write dequantized f16
static __device__ inline void fq_block16(const float* __restrict__ in, long long instride,
                                         _Float16* __restrict__ out, float gscale) {
  float v[16];
  float bamax = 0.f;
#pragma unroll
  for (int j = 0; j < 16; ++j) {
    v[j] = in[(long long)j * instride];
    bamax = fmaxf(bamax, fabsf(v[j]));
  }
  float sf  = quant_e4m3(bamax * (gscale * (1.f / 6.f)));
  float ssf = (sf > 0.f) ? sf : 1.f;
  float s   = gscale / ssf;
  float inv = sf / gscale;
#pragma unroll
  for (int j = 0; j < 16; ++j)
    out[j] = (_Float16)(round_fp4(v[j] * s) * inv);
}

// ---------- amax kernels ----------

__global__ void zero_amax_kernel(unsigned* __restrict__ amax) {
  if (threadIdx.x < 32) amax[threadIdx.x] = 0u;
}

// grid: (blocks, nslices); slice s covers x + s*nper, result atomicMax into amax_out[s]
__global__ __launch_bounds__(256) void amax_kernel(const float* __restrict__ x, long long nper,
                                                   unsigned* __restrict__ amax_out) {
  long long s = blockIdx.y;
  const float* p = x + s * nper;
  float m = 0.f;
  for (long long i = (long long)blockIdx.x * blockDim.x + threadIdx.x; i < nper;
       i += (long long)gridDim.x * blockDim.x)
    m = fmaxf(m, fabsf(p[i]));
  __shared__ float red[256];
  red[threadIdx.x] = m;
  __syncthreads();
  for (int st = 128; st > 0; st >>= 1) {
    if ((int)threadIdx.x < st) red[threadIdx.x] = fmaxf(red[threadIdx.x], red[threadIdx.x + st]);
    __syncthreads();
  }
  if (threadIdx.x == 0) atomicMax(amax_out + s, __float_as_uint(red[0]));
}

// ---------- quantization kernels ----------

__global__ __launch_bounds__(256) void quant_hs_kernel(const float* __restrict__ hs,
                                                       const unsigned* __restrict__ amax,
                                                       _Float16* __restrict__ hsq) {
  long long b = (long long)blockIdx.x * blockDim.x + threadIdx.x;   // T*H/16 blocks
  if (b >= (long long)Tt * Hh / 16) return;
  float am = fmaxf(__uint_as_float(amax[0]), 1e-12f);
  float gscale = 2688.f / am;                                       // 448*6/amax
  fq_block16(hs + b * 16, 1, hsq + b * 16, gscale);
}

// w13[e,o,h]: o<I -> up = gate_up_proj[e,h,2o+1]; o>=I -> gate = gate_up_proj[e,h,2(o-I)]
__global__ __launch_bounds__(256) void quant_w13_kernel(const float* __restrict__ gup,
                                                        const unsigned* __restrict__ amax,
                                                        _Float16* __restrict__ w13q) {
  long long b = (long long)blockIdx.x * blockDim.x + threadIdx.x;   // E*2I*(H/16)
  if (b >= (long long)Ee * 2 * Ii * (Hh / 16)) return;
  int hb = (int)(b % (Hh / 16));
  long long r = b / (Hh / 16);
  int o = (int)(r % (2 * Ii));
  int e = (int)(r / (2 * Ii));
  float am = fmaxf(__uint_as_float(amax[1 + e]), 1e-12f);
  float gscale = 2688.f / am;
  int c = (o < Ii) ? (2 * o + 1) : (2 * (o - Ii));
  const float* in = gup + ((long long)e * Hh + (long long)hb * 16) * (2LL * Ii) + c;
  _Float16* outp  = w13q + ((long long)e * 2 * Ii + o) * Hh + hb * 16;
  fq_block16(in, 2LL * Ii, outp, gscale);
}

// w2[e,h,i] = down_proj[e,i,h]
__global__ __launch_bounds__(256) void quant_w2_kernel(const float* __restrict__ dp,
                                                       const unsigned* __restrict__ amax,
                                                       _Float16* __restrict__ w2q) {
  long long b = (long long)blockIdx.x * blockDim.x + threadIdx.x;   // E*H*(I/16)
  if (b >= (long long)Ee * Hh * (Ii / 16)) return;
  int ib = (int)(b % (Ii / 16));
  long long r = b / (Ii / 16);
  int h = (int)(r % Hh);
  int e = (int)(r / Hh);
  float am = fmaxf(__uint_as_float(amax[9 + e]), 1e-12f);
  float gscale = 2688.f / am;
  const float* in = dp + ((long long)e * Ii + (long long)ib * 16) * Hh + h;
  _Float16* outp  = w2q + ((long long)e * Hh + h) * Ii + ib * 16;
  fq_block16(in, Hh, outp, gscale);
}

// ---------- fragment helpers ----------

static __device__ inline v16h ld_frag(const _Float16* __restrict__ base, int row, int khalf) {
  const v8h lo = *(const v8h*)(base + row * 40 + khalf);
  const v8h hi = *(const v8h*)(base + row * 40 + khalf + 16);
  return __builtin_shufflevector(lo, hi, 0,1,2,3,4,5,6,7,8,9,10,11,12,13,14,15);
}

// ---------- GEMM1: h1 = hsq @ w13q^T, fused bias + clip + gated SiLU -> act (E,T,I) f16 ----------
// grid (I/64, T/128, E), block 256 (8 wave32). Double-buffered LDS fed by async DMA.
__global__ __launch_bounds__(256) void gemm1_act_kernel(const _Float16* __restrict__ hsq,
                                                        const _Float16* __restrict__ w13q,
                                                        const float* __restrict__ gub,
                                                        _Float16* __restrict__ act) {
  const int e  = blockIdx.z;
  const int m0 = blockIdx.y * 128;
  const int n0 = blockIdx.x * 64;
  __shared__ __align__(16) _Float16 sA[2][128 * 40];
  __shared__ __align__(16) _Float16 sU[2][64 * 40];
  __shared__ __align__(16) _Float16 sG[2][64 * 40];
  const int tid  = threadIdx.x;
  const int lane = tid & 31;
  const int wave = tid >> 5;

  const _Float16* wU = w13q + ((long long)e * 2 * Ii + n0) * Hh;        // up rows
  const _Float16* wG = w13q + ((long long)e * 2 * Ii + Ii + n0) * Hh;   // gate rows
  const _Float16* xA = hsq + (long long)m0 * Hh;

  // per-thread DMA coordinates: row r0 (0..63) / r0+64 for A; row r0 for U/G; 8-half chunk kq
  const int r0 = tid >> 2, kq = tid & 3;
  const unsigned ldsOff = (unsigned)((r0 * 40 + kq * 8) * 2);
  const unsigned aB = lds_lo32(&sA[0][0]) + ldsOff;
  const unsigned uB = lds_lo32(&sU[0][0]) + ldsOff;
  const unsigned gB = lds_lo32(&sG[0][0]) + ldsOff;
  const long long gOff = (long long)r0 * Hh + kq * 8;

  v8f accU[4], accG[4];
#pragma unroll
  for (int i = 0; i < 4; ++i) { accU[i] = vzero8f(); accG[i] = vzero8f(); }

  // preload tile k0=0 into buffer 0
  async_ld_b128(aB,                     xA + gOff);
  async_ld_b128(aB + 64u * 40u * 2u,    xA + gOff + 64LL * Hh);
  async_ld_b128(uB,                     wU + gOff);
  async_ld_b128(gB,                     wG + gOff);

  int p = 0;
  for (int k0 = 0; k0 < Hh; k0 += 32, p ^= 1) {
    wait_async0();        // this wave's DMA into buf p complete
    __syncthreads();      // everyone's DMA complete; prior reads of buf p drained
    if (k0 + 32 < Hh) {   // kick next tile's DMA into buf p^1, overlapping compute
      const unsigned pb = (unsigned)(p ^ 1);
      async_ld_b128(aB + pb * (128u * 40u * 2u),                  xA + gOff + (k0 + 32));
      async_ld_b128(aB + pb * (128u * 40u * 2u) + 64u * 40u * 2u, xA + gOff + 64LL * Hh + (k0 + 32));
      async_ld_b128(uB + pb * (64u * 40u * 2u),                   wU + gOff + (k0 + 32));
      async_ld_b128(gB + pb * (64u * 40u * 2u),                   wG + gOff + (k0 + 32));
    }

    const _Float16* Ap = &sA[p][0];
    const _Float16* Up = &sU[p][0];
    const _Float16* Gp = &sG[p][0];
    const int khalf = (lane < 16) ? 0 : 8;  // lanes 0-15: K 0-7/16-23; lanes 16-31: K 8-15/24-31
    const v16h af = ld_frag(Ap, wave * 16 + (lane & 15), khalf);
    v16h bu[4], bg[4];
#pragma unroll
    for (int nt = 0; nt < 4; ++nt) {
      bu[nt] = ld_frag(Up, nt * 16 + (lane & 15), khalf);
      bg[nt] = ld_frag(Gp, nt * 16 + (lane & 15), khalf);
    }
    sched_fence();   // all fragment ds_loads issued (distinct regs) before the WMMA burst
#pragma unroll
    for (int nt = 0; nt < 4; ++nt) {
      accU[nt] = __builtin_amdgcn_wmma_f32_16x16x32_f16(false, af, false, bu[nt], (short)0, accU[nt], false, false);
      accG[nt] = __builtin_amdgcn_wmma_f32_16x16x32_f16(false, af, false, bg[nt], (short)0, accG[nt], false, false);
    }
    sched_fence();
  }

  // epilogue: bias + clip + gated SiLU, write act f16
  const int rbase = m0 + wave * 16 + ((lane < 16) ? 0 : 8);
#pragma unroll
  for (int nt = 0; nt < 4; ++nt) {
    const int col = n0 + nt * 16 + (lane & 15);
    const float bu = gub[e * 2 * Ii + 2 * col + 1];
    const float bg = gub[e * 2 * Ii + 2 * col];
#pragma unroll
    for (int j = 0; j < 8; ++j) {
      const int row = rbase + j;
      float up = accU[nt][j] + bu;
      float g  = accG[nt][j] + bg;
      g  = fminf(g, 7.f);
      up = fminf(fmaxf(up, -7.f), 7.f);
      const float sig = 1.f / (1.f + __expf(-1.702f * g));
      const float a   = (up + 1.f) * (g * sig);
      act[((long long)e * Tt + row) * Ii + col] = (_Float16)a;
    }
  }
}

// ---------- GEMM2: out[t,h] = sum_e combine[t,e]*(act[e]@w2[e]^T + b2[e]) ----------
// grid (H/64, T/128), block 256. Loops all 8 experts, accumulates in registers, writes once.
__global__ __launch_bounds__(256) void gemm2_out_kernel(const _Float16* __restrict__ act,
                                                        const _Float16* __restrict__ w2q,
                                                        const float* __restrict__ dpb,
                                                        const int* __restrict__ ridx,
                                                        const float* __restrict__ rw,
                                                        float* __restrict__ out) {
  const int m0 = blockIdx.y * 128;
  const int n0 = blockIdx.x * 64;
  __shared__ __align__(16) _Float16 sA[2][128 * 40];
  __shared__ __align__(16) _Float16 sB[2][64 * 40];
  __shared__ float comb[128][8];
  const int tid = threadIdx.x, lane = tid & 31, wave = tid >> 5;

  if (tid < 128) {  // combine[t,e] = routing_weights[t,e] * #{k: router_indices[t,k]==e}
    const int t = m0 + tid;
    const int i0 = ridx[t * 4 + 0], i1 = ridx[t * 4 + 1], i2 = ridx[t * 4 + 2], i3 = ridx[t * 4 + 3];
#pragma unroll
    for (int e = 0; e < 8; ++e) {
      const float cnt = (float)((i0 == e) + (i1 == e) + (i2 == e) + (i3 == e));
      comb[tid][e] = cnt * rw[t * 8 + e];
    }
  }

  const int r0 = tid >> 2, kq = tid & 3;
  const unsigned ldsOff = (unsigned)((r0 * 40 + kq * 8) * 2);
  const unsigned aB = lds_lo32(&sA[0][0]) + ldsOff;
  const unsigned bB = lds_lo32(&sB[0][0]) + ldsOff;
  const long long gOff = (long long)r0 * Ii + kq * 8;

  v8f fin[4];
#pragma unroll
  for (int i = 0; i < 4; ++i) fin[i] = vzero8f();

  for (int e = 0; e < 8; ++e) {
    v8f acc[4];
#pragma unroll
    for (int i = 0; i < 4; ++i) acc[i] = vzero8f();
    const _Float16* xA = act + ((long long)e * Tt + m0) * Ii;
    const _Float16* xB = w2q + ((long long)e * Hh + n0) * Ii;

    // preload tile k0=0 into buffer 0
    async_ld_b128(aB,                  xA + gOff);
    async_ld_b128(aB + 64u * 40u * 2u, xA + gOff + 64LL * Ii);
    async_ld_b128(bB,                  xB + gOff);

    int p = 0;
    for (int k0 = 0; k0 < Ii; k0 += 32, p ^= 1) {
      wait_async0();
      __syncthreads();
      if (k0 + 32 < Ii) {
        const unsigned pb = (unsigned)(p ^ 1);
        async_ld_b128(aB + pb * (128u * 40u * 2u),                  xA + gOff + (k0 + 32));
        async_ld_b128(aB + pb * (128u * 40u * 2u) + 64u * 40u * 2u, xA + gOff + 64LL * Ii + (k0 + 32));
        async_ld_b128(bB + pb * (64u * 40u * 2u),                   xB + gOff + (k0 + 32));
      }

      const _Float16* Ap = &sA[p][0];
      const _Float16* Bp = &sB[p][0];
      const int khalf = (lane < 16) ? 0 : 8;
      const v16h af = ld_frag(Ap, wave * 16 + (lane & 15), khalf);
      v16h bf[4];
#pragma unroll
      for (int nt = 0; nt < 4; ++nt) bf[nt] = ld_frag(Bp, nt * 16 + (lane & 15), khalf);
      sched_fence();   // ds_load burst before WMMA burst
#pragma unroll
      for (int nt = 0; nt < 4; ++nt)
        acc[nt] = __builtin_amdgcn_wmma_f32_16x16x32_f16(false, af, false, bf[nt], (short)0, acc[nt], false, false);
      sched_fence();
    }
    __syncthreads();  // buf reads done before next expert's preload overwrites

    const int rb = wave * 16 + ((lane < 16) ? 0 : 8);
#pragma unroll
    for (int nt = 0; nt < 4; ++nt) {
      const int col  = n0 + nt * 16 + (lane & 15);
      const float b2 = dpb[e * Hh + col];
#pragma unroll
      for (int j = 0; j < 8; ++j) {
        const float cw = comb[rb + j][e];
        fin[nt][j] += cw * (acc[nt][j] + b2);
      }
    }
  }

  const int rbase = m0 + wave * 16 + ((lane < 16) ? 0 : 8);
#pragma unroll
  for (int nt = 0; nt < 4; ++nt) {
    const int col = n0 + nt * 16 + (lane & 15);
#pragma unroll
    for (int j = 0; j < 8; ++j)
      out[(long long)(rbase + j) * Hh + col] = fin[nt][j];
  }
}

// ---------- launch ----------

extern "C" void kernel_launch(void* const* d_in, const int* in_sizes, int n_in,
                              void* d_out, int out_size, void* d_ws, size_t ws_size,
                              hipStream_t stream) {
  (void)in_sizes; (void)n_in; (void)out_size; (void)ws_size;
  const float* hs  = (const float*)d_in[0];
  const float* gup = (const float*)d_in[1];
  const float* gub = (const float*)d_in[2];
  const float* dp  = (const float*)d_in[3];
  const float* dpb = (const float*)d_in[4];
  const int*   rix = (const int*)d_in[5];
  const float* rw  = (const float*)d_in[6];
  float* out = (float*)d_out;

  char* ws = (char*)d_ws;
  unsigned* amax = (unsigned*)ws;                               // [0]=hs, [1..8]=w13/e, [9..16]=w2/e
  _Float16* hsq  = (_Float16*)(ws + 256);                       // T*H f16            (2,949,120 B)
  _Float16* w13q = (_Float16*)(ws + 256 + 2949120LL);           // E*2I*H f16       (265,420,800 B)
  _Float16* w2q  = (_Float16*)(ws + 256 + 2949120LL + 265420800LL);      // E*H*I  (132,710,400 B)
  _Float16* actb = (_Float16*)(ws + 256 + 2949120LL + 265420800LL + 132710400LL); // E*T*I f16

  zero_amax_kernel<<<1, 32, 0, stream>>>(amax);
  amax_kernel<<<dim3(128, 1), 256, 0, stream>>>(hs,  (long long)Tt * Hh,       amax + 0);
  amax_kernel<<<dim3(256, Ee), 256, 0, stream>>>(gup, (long long)Hh * 2 * Ii,  amax + 1);
  amax_kernel<<<dim3(256, Ee), 256, 0, stream>>>(dp,  (long long)Ii * Hh,      amax + 9);

  quant_hs_kernel <<<(Tt * Hh / 16 + 255) / 256, 256, 0, stream>>>(hs, amax, hsq);
  quant_w13_kernel<<<(int)(((long long)Ee * 2 * Ii * (Hh / 16)) / 256), 256, 0, stream>>>(gup, amax, w13q);
  quant_w2_kernel <<<(int)(((long long)Ee * Hh * (Ii / 16)) / 256), 256, 0, stream>>>(dp, amax, w2q);

  gemm1_act_kernel<<<dim3(Ii / 64, Tt / 128, Ee), 256, 0, stream>>>(hsq, w13q, gub, actb);
  gemm2_out_kernel<<<dim3(Hh / 64, Tt / 128), 256, 0, stream>>>(actb, w2q, dpb, rix, rw, out);
}
